// EncoderDecoderEmbeddings_11914239279437
// MI455X (gfx1250) — matrix operand order repaired and checked by reading, not compiled, not run on hardware
//
#include <hip/hip_runtime.h>
#include <hip/hip_bf16.h>
#include <cstdint>
#include <cstddef>

typedef _Float16 v16h  __attribute__((ext_vector_type(16)));
typedef _Float16 h8    __attribute__((ext_vector_type(8)));
typedef float    v8f   __attribute__((ext_vector_type(8)));
typedef unsigned int u32x4 __attribute__((ext_vector_type(4)));
typedef unsigned int u4  __attribute__((ext_vector_type(4)));
typedef int          i4  __attribute__((ext_vector_type(4)));
typedef int          i8v __attribute__((ext_vector_type(8)));

#define B_TOT 8192
#define E_    32
#define P_    320              // S*H*W = 16*4*5
#define KD    10240            // P_*E_
#define NP    112              // N=100 padded to 7*16
#define RS2   0.70710678118654752440f   // DH^-0.5, DH=2

#define KC    32               // GEMM K-chunk (one WMMA k-step)
#define LDR   40               // padded LDS row stride in f16 (32 + 8)
#define NCHUNK (KD / KC)       // 320

// workspace layout (bytes)
#define CST_OFF 0
#define BT_OFF  4096                          // Wd1^T fp16 [NP][KD]
#define H1_OFF  (BT_OFF + (size_t)NP*KD*2)    // h1 f32 [B][NP]
#define F_OFF   (H1_OFF + (size_t)B_TOT*NP*4) // f fp16 [B][KD]

#if __has_builtin(__builtin_amdgcn_tensor_load_to_lds)
#define HAVE_TDM 1
#else
#define HAVE_TDM 0
#endif

// ---------------------------------------------------------------------------
// Kernel 0: derive the collapsed-attention constants.
// cst[0..11]=alpha[a*4+h], cst[12..23]=gamma, cst[24..407]=wvec[(a*4+h)*32+e],
// cst[408..439]=Cvec[e]
// ---------------------------------------------------------------------------
__global__ void k_setup(const float* __restrict__ We, const float* __restrict__ be,
                        const float* __restrict__ Wq, const float* __restrict__ Wkv,
                        const float* __restrict__ Wo, const float* __restrict__ bo,
                        float* __restrict__ cst) {
  const int e = threadIdx.x;   // 0..31
  float cvec = 0.f;
  for (int a = 0; a < 3; ++a) {
    const float* wq  = Wq  + a * E_ * 8;
    const float* wkv = Wkv + a * E_ * 16;
    const float* wo  = Wo  + a * 8 * E_;
    float vA[8], vB[8];
    for (int c = 0; c < 8; ++c) {
      float sA = 0.f, sB = 0.f;
      for (int ee = 0; ee < E_; ++ee) {
        float wv = wkv[ee * 16 + 8 + c];     // v half of Wkv
        sA += We[ee] * wv;  sB += be[ee] * wv;
      }
      vA[c] = sA; vB[c] = sB;
    }
    for (int h = 0; h < 4; ++h) {
      float w0 = wo[(2*h)*E_ + e], w1 = wo[(2*h+1)*E_ + e];
      cst[24 + (a*4 + h)*E_ + e] = vA[2*h]*w0 + vA[2*h+1]*w1;
      cvec += vB[2*h]*w0 + vB[2*h+1]*w1;
    }
    cvec += bo[a*E_ + e];
    if (e == 0) {
      float qA[8], qB[8], kA[8];
      for (int c = 0; c < 8; ++c) {
        float sA=0.f, sB=0.f, sK=0.f;
        for (int ee = 0; ee < E_; ++ee) {
          sA += We[ee]*wq[ee*8 + c];
          sB += be[ee]*wq[ee*8 + c];
          sK += We[ee]*wkv[ee*16 + c];       // k half of Wkv
        }
        qA[c]=sA; qB[c]=sB; kA[c]=sK;
      }
      for (int h = 0; h < 4; ++h) {
        cst[a*4 + h]      = (qA[2*h]*kA[2*h] + qA[2*h+1]*kA[2*h+1]) * RS2;
        cst[12 + a*4 + h] = (qB[2*h]*kA[2*h] + qB[2*h+1]*kA[2*h+1]) * RS2;
      }
    }
  }
  cst[408 + e] = cvec;
}

// ---------------------------------------------------------------------------
// Kernel 0b: Wd1 [KD,100] f32 -> BT [NP][KD] f16 (transposed, zero-padded)
// ---------------------------------------------------------------------------
__global__ void k_wd1(const float* __restrict__ Wd1, _Float16* __restrict__ BT) {
  int idx = blockIdx.x * 256 + threadIdx.x;
  if (idx >= NP * KD) return;
  int n = idx / KD, k = idx - n * KD;
  float v = (n < 100) ? Wd1[k * 100 + n] : 0.f;
  BT[idx] = (_Float16)v;
}

// ---------------------------------------------------------------------------
// Kernel 1: collapsed axial attention -> f fp16 [B][KD]. One wave per batch row.
// ---------------------------------------------------------------------------
__global__ void __launch_bounds__(128) k_attn(const float* __restrict__ x,
                                              const float* __restrict__ cst,
                                              _Float16* __restrict__ F) {
  __shared__ float    u_s[4][P_];
  __shared__ _Float16 m_s[4][12][P_];
  const int tid = threadIdx.x, wave = tid >> 5, lane = tid & 31;
  const int b = blockIdx.x * 4 + wave;
  float* u = u_s[wave];
  const float* xr = x + (size_t)b * P_;
  for (int t = 0; t < 10; ++t) u[lane + t*32] = xr[lane + t*32];
  __syncthreads();
  _Float16 (*mm)[P_] = m_s[wave];

  // axis 0: attention along S (L=16); lines indexed by (h,w), p = i*20 + l
  { int li = lane >> 4, i = lane & 15;
    for (int lp = 0; lp < 10; ++lp) {
      int l = lp*2 + li;
      float ui = u[i*20 + l];
      for (int hh = 0; hh < 4; ++hh) {
        float c = cst[hh]*ui + cst[12 + hh];
        float den = 0.f, num = 0.f;
        for (int j = 0; j < 16; ++j) { float uj = u[j*20 + l]; float ex = __expf(c*uj); den += ex; num += ex*uj; }
        mm[hh][i*20 + l] = (_Float16)(num / den);
      } } }
  // axis 1: along H (L=4); lines (s,w), p = s*20 + i*5 + w
  { int li = lane >> 2, i = lane & 3;
    for (int lb = 0; lb < 10; ++lb) {
      int l = lb*8 + li; int s = l / 5, w = l - s*5; int pb = s*20 + w;
      float ui = u[pb + i*5];
      for (int hh = 0; hh < 4; ++hh) {
        float c = cst[4 + hh]*ui + cst[16 + hh];
        float den = 0.f, num = 0.f;
        for (int j = 0; j < 4; ++j) { float uj = u[pb + j*5]; float ex = __expf(c*uj); den += ex; num += ex*uj; }
        mm[4 + hh][pb + i*5] = (_Float16)(num / den);
      } } }
  // axis 2: along W (L=5); lines (s,h), p = s*20 + h*5 + i  (lanes 30,31 idle)
  { int li = lane / 5, i = lane - li*5; bool act = lane < 30;
    for (int lb = 0; lb < 11; ++lb) {
      int l = lb*6 + li; bool on = act && (l < 64); int lc = on ? l : 0;
      int s = lc >> 2, hx = lc & 3; int pb = s*20 + hx*5;
      float ui = u[pb + i];
      for (int hh = 0; hh < 4; ++hh) {
        float c = cst[8 + hh]*ui + cst[20 + hh];
        float den = 0.f, num = 0.f;
        for (int j = 0; j < 5; ++j) { float uj = u[pb + j]; float ex = __expf(c*uj); den += ex; num += ex*uj; }
        if (on) mm[8 + hh][pb + i] = (_Float16)(num / den);
      } } }
  __syncthreads();

  // phase B: f[p,e] = relu( Cvec[e] + sum_t m[t][p] * wvec[t][e] ); lane = e
  float wv[12];
  #pragma unroll
  for (int t = 0; t < 12; ++t) wv[t] = cst[24 + t*32 + lane];
  float cc = cst[408 + lane];
  _Float16* fo = F + (size_t)b * KD + lane;
  for (int p = 0; p < P_; ++p) {
    float acc = cc;
    #pragma unroll
    for (int t = 0; t < 12; ++t) acc += (float)mm[t][p] * wv[t];
    acc = fmaxf(acc, 0.f);
    fo[p * 32] = (_Float16)acc;
  }
}

// ---------------------------------------------------------------------------
// Kernel 2: WMMA GEMM  h1 = lrelu(f @ Wd1 + bd1)
// TDM (tensor_load_to_lds) double-buffered staging, K-chunk 32.
// ---------------------------------------------------------------------------
union AF { v16h v; struct { h8 lo; h8 hi; } p; };

__device__ __forceinline__ unsigned lds_off(const void* p) {
  // generic pointers to LDS carry the wave-relative byte offset in the low 32 bits
  return (unsigned)(uintptr_t)p;
}

#if HAVE_TDM
// 2D TDM tile load: tile_rows x KC f16 elements, row stride KD elements,
// LDS rows padded by 4 DWORDs (pad_interval=3: 16 DWORDs, pad_amount=3: 4 DWORDs)
__device__ __forceinline__ void tdm_load_2d(unsigned lds_addr, const _Float16* gptr,
                                            unsigned tile_rows) {
  unsigned long long ga = (unsigned long long)(uintptr_t)gptr;
  u4 g0;
  g0[0] = 1u;                                        // count=1 (valid), no gather
  g0[1] = lds_addr;                                  // LDS byte address
  g0[2] = (unsigned)(ga & 0xFFFFFFFFu);              // global_addr[31:0]
  g0[3] = (unsigned)((ga >> 32) & 0x01FFFFFFu) | (2u << 30);  // addr[56:32] | type=2
  i8v g1;
  g1[0] = (int)((1u << 16) |                         // data_size = 2B
                (1u << 20) |                         // pad_enable
                (3u << 22) |                         // pad_interval: 16 DWORDs (one row)
                (3u << 25));                         // pad_amount: 4 DWORDs
  unsigned td0 = 0x40000000u;                        // tensor_dim0 (huge: never OOB)
  unsigned td1 = 0x40000000u;                        // tensor_dim1
  g1[1] = (int)((td0 & 0xFFFFu) << 16);
  g1[2] = (int)(((td0 >> 16) & 0xFFFFu) | ((td1 & 0xFFFFu) << 16));
  g1[3] = (int)(((td1 >> 16) & 0xFFFFu) | ((unsigned)KC << 16));   // tile_dim0 = 32
  g1[4] = (int)(tile_rows & 0xFFFFu);                // tile_dim1; tile_dim2=0
  g1[5] = (int)KD;                                   // tensor_dim0_stride[31:0]
  g1[6] = 0;                                         // stride hi / dim1_stride lo
  g1[7] = 0;
  i4 g2; g2[0]=0; g2[1]=0; g2[2]=0; g2[3]=0;
  i4 g3; g3[0]=0; g3[1]=0; g3[2]=0; g3[3]=0;
#if defined(__clang_major__) && (__clang_major__ >= 23)
  i8v g4; g4[0]=0; g4[1]=0; g4[2]=0; g4[3]=0; g4[4]=0; g4[5]=0; g4[6]=0; g4[7]=0;
  __builtin_amdgcn_tensor_load_to_lds(g0, g1, g2, g3, g4, 0);
#else
  __builtin_amdgcn_tensor_load_to_lds(g0, g1, g2, g3, 0);
#endif
}
#endif

__device__ __forceinline__ void gemm_chunk(const _Float16* as, const _Float16* bs,
                                           int wave, int lane, v8f acc[2][7]) {
  AF af0, af1;
  const int rowb = wave*32 + (lane & 15);
  const int kb = (lane >> 4) * 8;        // A frag: K base..base+7 and base+16..base+23
  af0.p.lo = *(const h8*)(as + (rowb     )*LDR + kb);
  af0.p.hi = *(const h8*)(as + (rowb     )*LDR + kb + 16);
  af1.p.lo = *(const h8*)(as + (rowb + 16)*LDR + kb);
  af1.p.hi = *(const h8*)(as + (rowb + 16)*LDR + kb + 16);
  const int kb2 = (lane >> 4) * 8;       // B frag: K base..base+7 and base+16..base+23
  #pragma unroll
  for (int nt = 0; nt < 7; ++nt) {
    AF bf;
    int n = nt*16 + (lane & 15);
    bf.p.lo = *(const h8*)(bs + n*LDR + kb2);
    bf.p.hi = *(const h8*)(bs + n*LDR + kb2 + 16);
    acc[0][nt] = __builtin_amdgcn_wmma_f32_16x16x32_f16(
        false, af0.v, false, bf.v, (short)0, acc[0][nt], false, false);
    acc[1][nt] = __builtin_amdgcn_wmma_f32_16x16x32_f16(
        false, af1.v, false, bf.v, (short)0, acc[1][nt], false, false);
  }
}

__global__ void __launch_bounds__(128) k_gemm(const _Float16* __restrict__ F,
                                              const _Float16* __restrict__ BT,
                                              const float* __restrict__ bd1,
                                              float* __restrict__ H1) {
  __shared__ _Float16 As[2][128 * LDR];  // 2 x 10240 B
  __shared__ _Float16 Bs[2][NP  * LDR];  // 2 x  8960 B
  const int tid = threadIdx.x, wave = tid >> 5, lane = tid & 31;
  const int M0 = blockIdx.x * 128;
  v8f acc[2][7];
  #pragma unroll
  for (int mt = 0; mt < 2; ++mt)
    #pragma unroll
    for (int nt = 0; nt < 7; ++nt)
      #pragma unroll
      for (int q = 0; q < 8; ++q) acc[mt][nt][q] = 0.f;

  const _Float16* FA = F + (size_t)M0 * KD;

#if HAVE_TDM
  const unsigned aoff[2] = { lds_off(&As[0][0]), lds_off(&As[1][0]) };
  const unsigned boff[2] = { lds_off(&Bs[0][0]), lds_off(&Bs[1][0]) };
  if (wave == 0) {                                   // prologue DMA: chunk 0
    tdm_load_2d(aoff[0], FA, 128);
    tdm_load_2d(boff[0], BT, NP);
    __builtin_amdgcn_s_wait_tensorcnt((short)0);
  }
  __syncthreads();
  #pragma unroll 1
  for (int kc = 0; kc < NCHUNK; ++kc) {
    const int cur = kc & 1;
    if (wave == 0 && (kc + 1) < NCHUNK) {            // DMA next chunk into other buffer
      tdm_load_2d(aoff[1 - cur], FA + (kc + 1) * KC, 128);
      tdm_load_2d(boff[1 - cur], BT + (kc + 1) * KC, NP);
    }
    gemm_chunk(As[cur], Bs[cur], wave, lane, acc);   // 14 WMMAs on current buffer
    if (wave == 0) __builtin_amdgcn_s_wait_tensorcnt((short)0);
    __syncthreads();
  }
#else
  // fallback: manual staging (single buffer)
  #pragma unroll 1
  for (int kc = 0; kc < NCHUNK; ++kc) {
    const int k0 = kc * KC;
    {
      int r = tid >> 2, kq = tid & 3;                // A: 128 rows x 32 f16
      #pragma unroll
      for (int it = 0; it < 4; ++it) {
        int rr = r + it * 32;
        *(u32x4*)(&As[0][rr*LDR + kq*8]) =
            *(const u32x4*)(FA + (size_t)rr*KD + k0 + kq*8);
      }
      for (int idx = tid; idx < NP*4; idx += 128) {  // B: 112 rows x 32 f16
        int n = idx >> 2, kq2 = idx & 3;
        *(u32x4*)(&Bs[0][n*LDR + kq2*8]) =
            *(const u32x4*)(BT + (size_t)n*KD + k0 + kq2*8);
      }
    }
    __syncthreads();
    if (kc + 1 < NCHUNK) {
      __builtin_prefetch(FA + (size_t)(tid >> 2)*KD + k0 + KC + (tid & 3)*8, 0, 3);
      __builtin_prefetch(BT + (size_t)(tid >> 2)*KD + k0 + KC + (tid & 3)*8, 0, 3);
    }
    gemm_chunk(As[0], Bs[0], wave, lane, acc);
    __syncthreads();
  }
#endif

  // epilogue: bias + LeakyReLU(0.2); C/D layout: lane%16 = N, VGPR r + (lane/16)*8 = M
  #pragma unroll
  for (int mt = 0; mt < 2; ++mt) {
    int rbase = M0 + wave*32 + mt*16 + (lane >> 4)*8;
    #pragma unroll
    for (int nt = 0; nt < 7; ++nt) {
      int n = nt*16 + (lane & 15);
      float bias = (n < 100) ? bd1[n] : 0.f;
      #pragma unroll
      for (int r = 0; r < 8; ++r) {
        float z = acc[mt][nt][r] + bias;
        z = (z >= 0.f) ? z : 0.2f * z;
        H1[(size_t)(rbase + r)*NP + n] = z;
      }
    }
  }
}

// ---------------------------------------------------------------------------
// Kernel 3: h2 = lrelu(h1 @ Wd2 + bd2); out = sigmoid(h2 @ Wd3 + bd3)
// ---------------------------------------------------------------------------
__global__ void __launch_bounds__(128) k_mlp(const float* __restrict__ H1,
                                             const float* __restrict__ Wd2,
                                             const float* __restrict__ bd2,
                                             const float* __restrict__ Wd3,
                                             const float* __restrict__ bd3,
                                             float* __restrict__ out) {
  __shared__ float w2[100 * 64];
  __shared__ float b2[64];
  __shared__ float w3[64];
  const int tid = threadIdx.x;
  for (int idx = tid; idx < 6400; idx += 128) w2[idx] = Wd2[idx];
  if (tid < 64) { b2[tid] = bd2[tid]; w3[tid] = Wd3[tid]; }
  __syncthreads();
  const int row = blockIdx.x * 128 + tid;
  float r[100];
  const float* hr = H1 + (size_t)row * NP;
  #pragma unroll
  for (int k = 0; k < 100; ++k) r[k] = hr[k];
  float acc = 0.f;
  #pragma unroll 1
  for (int j = 0; j < 64; ++j) {
    float s = b2[j];
    #pragma unroll
    for (int k = 0; k < 100; ++k) s += r[k] * w2[k*64 + j];
    s = (s >= 0.f) ? s : 0.2f * s;
    acc += s * w3[j];
  }
  float z = acc + bd3[0];
  out[row] = 1.f / (1.f + __expf(-z));
}

// ---------------------------------------------------------------------------
extern "C" void kernel_launch(void* const* d_in, const int* in_sizes, int n_in,
                              void* d_out, int out_size, void* d_ws, size_t ws_size,
                              hipStream_t stream) {
  (void)in_sizes; (void)n_in; (void)out_size; (void)ws_size;
  const float* x    = (const float*)d_in[0];
  const float* We   = (const float*)d_in[1];
  const float* be   = (const float*)d_in[2];
  const float* Wq   = (const float*)d_in[3];
  const float* Wkv  = (const float*)d_in[4];
  const float* Wo   = (const float*)d_in[5];
  const float* bo   = (const float*)d_in[6];
  const float* Wd1  = (const float*)d_in[7];
  const float* bd1  = (const float*)d_in[8];
  const float* Wd2  = (const float*)d_in[9];
  const float* bd2  = (const float*)d_in[10];
  const float* Wd3  = (const float*)d_in[11];
  const float* bd3  = (const float*)d_in[12];
  char* ws = (char*)d_ws;
  float*    cst = (float*)(ws + CST_OFF);
  _Float16* BT  = (_Float16*)(ws + BT_OFF);
  float*    H1  = (float*)(ws + H1_OFF);
  _Float16* F   = (_Float16*)(ws + F_OFF);
  float*    out = (float*)d_out;

  hipLaunchKernelGGL(k_setup, dim3(1), dim3(32), 0, stream, We, be, Wq, Wkv, Wo, bo, cst);
  hipLaunchKernelGGL(k_wd1,  dim3((NP*KD + 255)/256), dim3(256), 0, stream, Wd1, BT);
  hipLaunchKernelGGL(k_attn, dim3(B_TOT/4),   dim3(128), 0, stream, x, cst, F);
  hipLaunchKernelGGL(k_gemm, dim3(B_TOT/128), dim3(128), 0, stream, F, BT, bd1, H1);
  hipLaunchKernelGGL(k_mlp,  dim3(B_TOT/128), dim3(128), 0, stream, H1, Wd2, bd2, Wd3, bd3, out);
}
